// FragmentEmbeddingToExpression_39006892982600
// MI455X (gfx1250) — matrix operand order repaired and checked by reading, not compiled, not run on hardware
//
#include <hip/hip_runtime.h>

typedef __attribute__((ext_vector_type(16))) _Float16 v16h;
typedef __attribute__((ext_vector_type(8)))  _Float16 v8h;
typedef __attribute__((ext_vector_type(8)))  float    v8f;
typedef __attribute__((ext_vector_type(4)))  float    v4f;

#define NCOMP   100
#define NKT     4      // K padded to 128 = 4 tiles of 32
#define NNT     7      // N padded to 112 = 7 tiles of 16
#define CELL_N  2000
#define GENE_N  1000
#define N_FRAG  2000000

// packed sizes (in _Float16 elements)
#define W12P_ELEMS (NNT * NKT * 32 * 16)   // 14336
#define W3P_ELEMS  (NKT * 32 * 16)         // 2048
#define PACK_TOTAL (2 * W12P_ELEMS + W3P_ELEMS)

// ---------------------------------------------------------------------------
// Pack W1/W2/W3 (f32, row-major [out,in]) into WMMA B-operand f16 layout:
// element index = ((nt*NKT + kt)*32 + lane)*16 + h
// B[k][n] with k = kt*32 + (h<8 ? 8*(lane/16)+h : 16+8*(lane/16)+(h-8)), n = nt*16 + lane%16
// B = W^T  =>  B[k][n] = W[n*100 + k]; zero-pad k>=100, n>=100.
// ---------------------------------------------------------------------------
__global__ void pack_weights_kernel(const float* __restrict__ W1,
                                    const float* __restrict__ W2,
                                    const float* __restrict__ W3,
                                    _Float16* __restrict__ W1p,
                                    _Float16* __restrict__ W2p,
                                    _Float16* __restrict__ W3p) {
    int i = blockIdx.x * blockDim.x + threadIdx.x;
    if (i >= PACK_TOTAL) return;
    const float* src;
    _Float16* dst;
    int li;
    bool isW3 = false;
    if (i < W12P_ELEMS)            { src = W1; dst = W1p; li = i; }
    else if (i < 2 * W12P_ELEMS)   { src = W2; dst = W2p; li = i - W12P_ELEMS; }
    else                           { src = W3; dst = W3p; li = i - 2 * W12P_ELEMS; isW3 = true; }

    int h    = li & 15;
    int lane = (li >> 4) & 31;
    int tile = li >> 9;                    // (nt*NKT + kt), or kt for W3
    int kt   = isW3 ? tile : (tile & 3);
    int nt   = isW3 ? 0    : (tile >> 2);
    int klocal = (h < 8) ? ((lane >> 4) * 8 + h) : (8 + (lane >> 4) * 8 + h);
    int k = kt * 32 + klocal;
    int n = nt * 16 + (lane & 15);

    float v = 0.0f;
    if (isW3) {
        if (k < NCOMP && n == 0) v = src[k];          // W3 is [1,100]
    } else {
        if (k < NCOMP && n < NCOMP) v = src[n * NCOMP + k];
    }
    dst[li] = (_Float16)v;
}

// ---------------------------------------------------------------------------
// out[c*GENE_N + g] = bias1[genes_oi[g]]  (also resets accumulator each call)
// ---------------------------------------------------------------------------
__global__ void init_out_kernel(const float* __restrict__ bias1,
                                const int* __restrict__ genes_oi,
                                float* __restrict__ out) {
    int i = blockIdx.x * blockDim.x + threadIdx.x;
    if (i >= CELL_N * GENE_N) return;
    int g = i % GENE_N;
    out[i] = bias1[genes_oi[g]];
}

// ---------------------------------------------------------------------------
// Main fused kernel: per-wave 16-fragment MLP via WMMA f16 + atomic scatter.
// ---------------------------------------------------------------------------
__global__ __launch_bounds__(256) void frag_mlp_scatter_kernel(
    const float* __restrict__ E,
    const float* __restrict__ b1, const float* __restrict__ b2,
    const float* __restrict__ b3,
    const int* __restrict__ ix,
    const _Float16* __restrict__ W1p, const _Float16* __restrict__ W2p,
    const _Float16* __restrict__ W3p,
    float* __restrict__ out)
{
    __shared__ _Float16 lds[8][16 * 128];   // per-wave 16x128 f16 activation scratch

    const int tid    = threadIdx.x;
    const int wave   = tid >> 5;
    const int lane   = tid & 31;
    const int lh     = lane & 15;           // 0..15
    const int hiHalf = lane >> 4;           // 0 or 1
    _Float16* scr = &lds[wave][0];

    // zero the pad columns 100..127 once (never rewritten)
    for (int i = lane; i < 16 * 28; i += 32) {
        int r = i / 28, c = 100 + (i % 28);
        scr[r * 128 + c] = (_Float16)0.0f;
    }
    __syncthreads();

    const int rowBase = (blockIdx.x * 8 + wave) * 16;
    const int row     = rowBase + lh;                 // A-operand: lane owns row lh
    const float* Erow = E + (long)row * NCOMP;

    // ---- layer 1: load A tiles straight from global (f32 -> f16) ----
    v16h a[NKT];
#pragma unroll
    for (int kt = 0; kt < NKT; ++kt) {
        v16h t;
        int kb = kt * 32 + hiHalf * 8;                // halves 0..7 -> K=kb..kb+7
        if (kt < 3) {                                 // fully in-range: vector loads
            v4f f0 = *(const v4f*)(Erow + kb);
            v4f f1 = *(const v4f*)(Erow + kb + 4);
            v4f f2 = *(const v4f*)(Erow + kb + 16);   // halves 8..15 -> K=kb+16..+23
            v4f f3 = *(const v4f*)(Erow + kb + 20);
#pragma unroll
            for (int j = 0; j < 4; ++j) {
                t[j]      = (_Float16)f0[j];
                t[4 + j]  = (_Float16)f1[j];
                t[8 + j]  = (_Float16)f2[j];
                t[12 + j] = (_Float16)f3[j];
            }
        } else {                                      // tail tile: guarded scalar
#pragma unroll
            for (int h = 0; h < 16; ++h) {
                int k = kb + ((h < 8) ? h : (8 + h));
                t[h] = (k < NCOMP) ? (_Float16)Erow[k] : (_Float16)0.0f;
            }
        }
        a[kt] = t;
    }

    v8f c[NNT];
#pragma unroll
    for (int nt = 0; nt < NNT; ++nt) c[nt] = v8f{};
#pragma unroll
    for (int kt = 0; kt < NKT; ++kt)
#pragma unroll
        for (int nt = 0; nt < NNT; ++nt) {
            v16h b = *(const v16h*)(W1p + (((nt * NKT) + kt) * 32 + lane) * 16);
            c[nt] = __builtin_amdgcn_wmma_f32_16x16x32_f16(
                false, a[kt], false, b, (short)0, c[nt], false, false);
        }

    // bias + relu, D-layout -> scratch (row-major 16x128 f16)
#pragma unroll
    for (int nt = 0; nt < NNT; ++nt) {
        int n = nt * 16 + lh;
        if (n < NCOMP) {
            float bias = b1[n];
#pragma unroll
            for (int r = 0; r < 8; ++r) {
                float v = c[nt][r] + bias;
                v = v > 0.0f ? v : 0.0f;
                int m = r + 8 * hiHalf;               // D: lanes 16-31 -> M=8..15
                scr[m * 128 + n] = (_Float16)v;
            }
        }
    }
    __syncthreads();

    // ---- layer 2: A from scratch (cross-lane transpose via LDS) ----
#pragma unroll
    for (int kt = 0; kt < NKT; ++kt) {
        int off = lh * 128 + kt * 32 + hiHalf * 8;
        v8h lo = *(const v8h*)(scr + off);
        v8h hi = *(const v8h*)(scr + off + 16);
        v16h t;
#pragma unroll
        for (int j = 0; j < 8; ++j) { t[j] = lo[j]; t[8 + j] = hi[j]; }
        a[kt] = t;
    }
#pragma unroll
    for (int nt = 0; nt < NNT; ++nt) c[nt] = v8f{};
#pragma unroll
    for (int kt = 0; kt < NKT; ++kt)
#pragma unroll
        for (int nt = 0; nt < NNT; ++nt) {
            v16h b = *(const v16h*)(W2p + (((nt * NKT) + kt) * 32 + lane) * 16);
            c[nt] = __builtin_amdgcn_wmma_f32_16x16x32_f16(
                false, a[kt], false, b, (short)0, c[nt], false, false);
        }
#pragma unroll
    for (int nt = 0; nt < NNT; ++nt) {
        int n = nt * 16 + lh;
        if (n < NCOMP) {
            float bias = b2[n];
#pragma unroll
            for (int r = 0; r < 8; ++r) {
                float v = c[nt][r] + bias;
                v = v > 0.0f ? v : 0.0f;
                int m = r + 8 * hiHalf;
                scr[m * 128 + n] = (_Float16)v;
            }
        }
    }
    __syncthreads();

    // ---- layer 3: single N-tile, only column 0 of B nonzero (= W3) ----
    v8f c3 = v8f{};
#pragma unroll
    for (int kt = 0; kt < NKT; ++kt) {
        int off = lh * 128 + kt * 32 + hiHalf * 8;
        v8h lo = *(const v8h*)(scr + off);
        v8h hi = *(const v8h*)(scr + off + 16);
        v16h t;
#pragma unroll
        for (int j = 0; j < 8; ++j) { t[j] = lo[j]; t[8 + j] = hi[j]; }
        v16h b = *(const v16h*)(W3p + (kt * 32 + lane) * 16);
        c3 = __builtin_amdgcn_wmma_f32_16x16x32_f16(
            false, t, false, b, (short)0, c3, false, false);
    }

    // column N=0 of D lives in lanes 0 (M=0..7) and 16 (M=8..15)
    if (lh == 0) {
        float bb = b3[0];
        int m0 = 8 * hiHalf;
#pragma unroll
        for (int r = 0; r < 8; ++r) {
            int frag = rowBase + m0 + r;
            int seg  = ix[frag];
            __hip_atomic_fetch_add(out + seg, c3[r] + bb,
                                   __ATOMIC_RELAXED, __HIP_MEMORY_SCOPE_AGENT);
        }
    }
}

// ---------------------------------------------------------------------------
extern "C" void kernel_launch(void* const* d_in, const int* in_sizes, int n_in,
                              void* d_out, int out_size, void* d_ws, size_t ws_size,
                              hipStream_t stream) {
    const float* E        = (const float*)d_in[0];
    const float* W1       = (const float*)d_in[1];
    const float* b1       = (const float*)d_in[2];
    const float* W2       = (const float*)d_in[3];
    const float* b2       = (const float*)d_in[4];
    const float* W3       = (const float*)d_in[5];
    const float* b3       = (const float*)d_in[6];
    const float* bias1    = (const float*)d_in[7];
    const int*   ix       = (const int*)d_in[8];
    const int*   genes_oi = (const int*)d_in[9];
    float* out = (float*)d_out;

    _Float16* ws  = (_Float16*)d_ws;
    _Float16* W1p = ws;
    _Float16* W2p = ws + W12P_ELEMS;
    _Float16* W3p = ws + 2 * W12P_ELEMS;

    pack_weights_kernel<<<(PACK_TOTAL + 255) / 256, 256, 0, stream>>>(
        W1, W2, W3, W1p, W2p, W3p);

    init_out_kernel<<<(CELL_N * GENE_N + 255) / 256, 256, 0, stream>>>(
        bias1, genes_oi, out);

    frag_mlp_scatter_kernel<<<N_FRAG / 128, 256, 0, stream>>>(
        E, b1, b2, b3, ix, W1p, W2p, W3p, out);
}
